// FOX_HPNS_RNNAgent_52484500357299
// MI455X (gfx1250) — matrix operand order repaired and checked by reading, not compiled, not run on hardware
//
#include <hip/hip_runtime.h>
#include <hip/hip_bf16.h>

// ---------- types ----------
typedef __attribute__((ext_vector_type(16))) _Float16 v16h;
typedef __attribute__((ext_vector_type(8)))  _Float16 v8h;
typedef __attribute__((ext_vector_type(8)))  float    v8f;

// ---------- problem constants ----------
constexpr int BSZ  = 2048;
constexpr int NAG  = 8;
constexpr int NEN  = 8;
constexpr int NAL  = 7;
constexpr int H    = 64;
constexpr int OUTN = 6;
constexpr int NACT = 14;          // 6 + 8
constexpr int Btot = BSZ * NAG;   // 16384
constexpr int RE   = Btot * NEN;  // 131072 enemy rows
constexpr int RA   = Btot * NAL;  // 114688 ally rows
constexpr int KCAT = 1056;        // 512(Ge)+512(Ga)+8(Se)+8(Sa)+16 pad
constexpr int HEOUT = 1154;

// ---------- workspace layout (bytes) ----------
constexpr size_t alignUp(size_t x) { return (x + 255) & ~size_t(255); }
constexpr size_t OFF_HIDE = 0;                                          // RE*64 f32
constexpr size_t OFF_AWS  = alignUp(OFF_HIDE + (size_t)RE * 64 * 4);    // Btot*KCAT f16
constexpr size_t OFF_WCAT = alignUp(OFF_AWS  + (size_t)Btot * KCAT * 2);// 128*KCAT f16
constexpr size_t OFF_WG1I = alignUp(OFF_WCAT + (size_t)128 * KCAT * 2);
constexpr size_t OFF_WG1H = alignUp(OFF_WG1I + (size_t)192 * 64 * 2);
constexpr size_t OFF_WG2I = alignUp(OFF_WG1H + (size_t)192 * 64 * 2);
constexpr size_t OFF_WG2H = alignUp(OFF_WG2I + (size_t)192 * 64 * 2);
constexpr size_t OFF_WE1  = alignUp(OFF_WG2H + (size_t)192 * 64 * 2);   // enc_W1^T [128][64]
constexpr size_t OFF_WE2  = alignUp(OFF_WE1  + (size_t)128 * 64 * 2);   // enc_W2^T [64][128]
constexpr size_t OFF_EMB  = alignUp(OFF_WE2  + (size_t)64 * 128 * 2);   // Btot*128 f32
constexpr size_t OFF_XH   = alignUp(OFF_EMB  + (size_t)Btot * 128 * 4); // x  f16
constexpr size_t OFF_X2H  = alignUp(OFF_XH   + (size_t)Btot * 64 * 2);  // x2 f16
constexpr size_t OFF_HFH  = alignUp(OFF_X2H  + (size_t)Btot * 64 * 2);  // hidden f16
constexpr size_t OFF_GI1  = alignUp(OFF_HFH  + (size_t)Btot * 64 * 2);  // Btot*192 f32
constexpr size_t OFF_GH1  = alignUp(OFF_GI1  + (size_t)Btot * 192 * 4);
constexpr size_t OFF_GI2  = alignUp(OFF_GH1  + (size_t)Btot * 192 * 4);
constexpr size_t OFF_GH2  = alignUp(OFF_GI2  + (size_t)Btot * 192 * 4);
constexpr size_t OFF_G    = alignUp(OFF_GH2  + (size_t)Btot * 192 * 4); // g f32
constexpr size_t OFF_GH16 = alignUp(OFF_G    + (size_t)Btot * 64 * 4);  // g f16
constexpr size_t OFF_Z1H  = alignUp(OFF_GH16 + (size_t)Btot * 64 * 2);  // z1 f16 (B*128)
constexpr size_t OFF_Z    = alignUp(OFF_Z1H  + (size_t)Btot * 128 * 2); // z f32
constexpr size_t OFF_U    = alignUp(OFF_Z    + (size_t)Btot * 64 * 4);  // u[b,2,64] f32
constexpr size_t OFF_S    = alignUp(OFF_U    + (size_t)Btot * 128 * 4); // s[b,2]
constexpr size_t OFF_QATT = alignUp(OFF_S    + (size_t)Btot * 2 * 4);   // B*8
constexpr size_t OFF_LQ   = alignUp(OFF_QATT + (size_t)Btot * 8 * 4);   // 112 f32

// =====================================================================
// K0: convert/transpose weights to f16 [N][K] layouts + build Wcat
// =====================================================================
__global__ void k0_prep(const float* __restrict__ heW2, const float* __restrict__ heb2,
                        const float* __restrict__ haW2, const float* __restrict__ hab2,
                        const float* __restrict__ g1i,  const float* __restrict__ g1h,
                        const float* __restrict__ g2i,  const float* __restrict__ g2h,
                        const float* __restrict__ e1,   const float* __restrict__ e2,
                        _Float16* __restrict__ Wcat,
                        _Float16* __restrict__ Wg1i, _Float16* __restrict__ Wg1h,
                        _Float16* __restrict__ Wg2i, _Float16* __restrict__ Wg2h,
                        _Float16* __restrict__ We1,  _Float16* __restrict__ We2) {
  int i = blockIdx.x * blockDim.x + threadIdx.x;
  int stride = gridDim.x * blockDim.x;
  for (int x = i; x < 128 * KCAT; x += stride) {
    int n = x / KCAT, c = x % KCAT;
    float v;
    if (c < 512)       { int j = c >> 3, d = c & 7;  v = heW2[(size_t)j * HEOUT + d * 128 + n]; }
    else if (c < 1024) { int c2 = c - 512; int j = c2 >> 3, d = c2 & 7;
                         v = haW2[(size_t)j * 1024 + d * 128 + n]; }
    else if (c < 1032) { v = heb2[(c - 1024) * 128 + n]; }
    else if (c < 1040) { v = hab2[(c - 1032) * 128 + n]; }
    else v = 0.0f;
    Wcat[x] = (_Float16)v;
  }
  for (int x = i; x < 192 * 64; x += stride) {
    int n = x >> 6, k = x & 63;
    Wg1i[x] = (_Float16)g1i[k * 192 + n];
    Wg1h[x] = (_Float16)g1h[k * 192 + n];
    Wg2i[x] = (_Float16)g2i[k * 192 + n];
    Wg2h[x] = (_Float16)g2h[k * 192 + n];
  }
  for (int x = i; x < 128 * 64; x += stride) { int n = x >> 6, k = x & 63;  We1[x] = (_Float16)e1[k * 128 + n]; }
  for (int x = i; x < 64 * 128; x += stride) { int n = x >> 7, k = x & 127; We2[x] = (_Float16)e2[k * 64 + n]; }
}

// =====================================================================
// K1e: enemy hyper layer-1 (relu) + Ge/Se build. 16 agents / block.
// =====================================================================
__global__ __launch_bounds__(256) void k1_enemy(const float* __restrict__ ef,
                                                const float* __restrict__ heW1,
                                                const float* __restrict__ heb1,
                                                float* __restrict__ hid_e,
                                                _Float16* __restrict__ Aws) {
  __shared__ float s_ef[128][8];
  __shared__ float s_hid[128][64];
  int t = threadIdx.x;
  int a0  = blockIdx.x * 16;
  int er0 = a0 * 8;
  for (int i = t; i < 128 * 8; i += 256) s_ef[i >> 3][i & 7] = ef[(size_t)er0 * 8 + i];
  __syncthreads();
  {
    int row = t >> 1, jb = (t & 1) * 32;
    for (int jj = 0; jj < 32; ++jj) {
      int j = jb + jj;
      float s = heb1[j];
      for (int d = 0; d < 8; ++d) s += s_ef[row][d] * heW1[d * 64 + j];
      s = fmaxf(s, 0.0f);
      s_hid[row][j] = s;
      hid_e[(size_t)(er0 + row) * 64 + j] = s;
    }
  }
  __syncthreads();
  int al = t >> 4, ob = (t & 15) * 32;
  for (int oo = 0; oo < 32; ++oo) {
    int o = ob + oo, j = o >> 3, d = o & 7;
    float s = 0.0f;
    for (int e = 0; e < 8; ++e) s += s_hid[al * 8 + e][j] * s_ef[al * 8 + e][d];
    Aws[(size_t)(a0 + al) * KCAT + o] = (_Float16)s;
  }
  if (t < 128) {
    int a2 = t >> 3, d = t & 7;
    float s = 0.0f;
    for (int e = 0; e < 8; ++e) s += s_ef[a2 * 8 + e][d];
    Aws[(size_t)(a0 + a2) * KCAT + 1024 + d] = (_Float16)s;
  }
  { int row = t >> 4, cc = t & 15;
    Aws[(size_t)(a0 + row) * KCAT + 1040 + cc] = (_Float16)0.0f; }
}

// =====================================================================
// K1a: ally hyper layer-1 (relu) + Ga/Sa build. 16 agents = 112 rows.
// =====================================================================
__global__ __launch_bounds__(256) void k1_ally(const float* __restrict__ af,
                                               const float* __restrict__ haW1,
                                               const float* __restrict__ hab1,
                                               _Float16* __restrict__ Aws) {
  __shared__ float s_af[112][8];
  __shared__ float s_hid[112][64];
  int t = threadIdx.x;
  int a0  = blockIdx.x * 16;
  int ar0 = a0 * 7;
  for (int i = t; i < 112 * 8; i += 256) s_af[i >> 3][i & 7] = af[(size_t)ar0 * 8 + i];
  __syncthreads();
  for (int idx = t; idx < 112 * 64; idx += 256) {
    int row = idx >> 6, j = idx & 63;
    float s = hab1[j];
    for (int d = 0; d < 8; ++d) s += s_af[row][d] * haW1[d * 64 + j];
    s_hid[row][j] = fmaxf(s, 0.0f);
  }
  __syncthreads();
  for (int idx = t; idx < 16 * 512; idx += 256) {
    int al = idx >> 9, o = idx & 511, j = o >> 3, d = o & 7;
    float s = 0.0f;
    for (int e = 0; e < 7; ++e) s += s_hid[al * 7 + e][j] * s_af[al * 7 + e][d];
    Aws[(size_t)(a0 + al) * KCAT + 512 + o] = (_Float16)s;
  }
  if (t < 128) {
    int al = t >> 3, d = t & 7;
    float s = 0.0f;
    for (int e = 0; e < 7; ++e) s += s_af[al * 7 + e][d];
    Aws[(size_t)(a0 + al) * KCAT + 1032 + d] = (_Float16)s;
  }
}

// =====================================================================
// Generic WMMA GEMM: C = act(A[M,K]_f16 @ Wt[N,K]_f16^T + bias), acc f32.
// One wave per 16x16 tile; A/B per-lane loads use one k order for gfx1250 wave32
// V_WMMA_F32_16X16X32_F16 operands (A: K run g*16+[0..15],
// B: K run g*16+[0..15], col/row = lane&15).
// =====================================================================
__global__ __launch_bounds__(256) void gemm_wmma_f16(const _Float16* __restrict__ A,
                                                     const _Float16* __restrict__ Wt,
                                                     const float* __restrict__ bias,
                                                     float* __restrict__ Cf,
                                                     _Float16* __restrict__ Ch,
                                                     int M, int N, int K, int relu) {
  int wave = blockIdx.x * (blockDim.x >> 5) + (threadIdx.x >> 5);
  int lane = threadIdx.x & 31;
  int nt = N >> 4;
  int r = wave / nt, c = wave % nt;
  if (r * 16 >= M) return;
  int g = lane >> 4, ln = lane & 15;
  v8f acc = {};
  const _Float16* arow = A  + (size_t)(r * 16 + ln) * K + g * 16;
  const _Float16* brow = Wt + (size_t)(c * 16 + ln) * K + g * 16;
  for (int kk = 0; kk < K; kk += 32) {
    v8h a0 = *(const v8h*)(arow + kk);
    v8h a1 = *(const v8h*)(arow + kk + 8);
    v16h b = *(const v16h*)(brow + kk);
    v16h a = __builtin_shufflevector(a0, a1, 0, 1, 2, 3, 4, 5, 6, 7,
                                             8, 9, 10, 11, 12, 13, 14, 15);
    acc = __builtin_amdgcn_wmma_f32_16x16x32_f16(false, a, false, b,
                                                 (short)0, acc, false, false);
  }
  int col = c * 16 + ln;
  float bv = bias ? bias[col] : 0.0f;
  for (int v = 0; v < 8; ++v) {
    int row = r * 16 + v + 8 * g;
    float val = acc[v] + bv;
    if (relu) val = fmaxf(val, 0.0f);
    if (Cf) Cf[(size_t)row * N + col] = val;
    if (Ch) Ch[(size_t)row * N + col] = (_Float16)val;
  }
}

// =====================================================================
// K3: per-agent fuse: fc1_own + embeddings + merger softmax + relu -> x;
//     fc1b + relu -> x2; hidden -> f16. 4 agents / 256-thread block.
// =====================================================================
__global__ void k3_agent(const float* __restrict__ own, const int* __restrict__ aids,
                         const int* __restrict__ lids, const float* __restrict__ hidden,
                         const float* __restrict__ fc1W, const float* __restrict__ fc1b,
                         const float* __restrict__ aemb, const float* __restrict__ acemb,
                         const float* __restrict__ mwin, const float* __restrict__ embsum,
                         const float* __restrict__ fc1bW, const float* __restrict__ fc1bb,
                         _Float16* __restrict__ xh, _Float16* __restrict__ x2h,
                         _Float16* __restrict__ hfh) {
  int t = threadIdx.x;
  int b = blockIdx.x * 4 + (t >> 6);
  int k = t & 63;
  const float* o = own + (size_t)b * 30;
  float e1 = fc1b[k], e2 = fc1bb[k];
  for (int i = 0; i < 30; ++i) {
    float ov = o[i];
    e1 += ov * fc1W[i * 64 + k];
    e2 += ov * fc1bW[i * 64 + k];
  }
  e1 += aemb[aids[b] * 64 + k] + acemb[lids[b] * 64 + k];
  float m0 = mwin[k], m1 = mwin[64 + k];
  float mx = fmaxf(m0, m1);
  float w0 = __expf(m0 - mx), w1 = __expf(m1 - mx);
  float inv = 1.0f / (w0 + w1); w0 *= inv; w1 *= inv;
  float merged = w0 * embsum[(size_t)b * 128 + k] + w1 * embsum[(size_t)b * 128 + 64 + k];
  xh [(size_t)b * 64 + k] = (_Float16)fmaxf(e1 + merged, 0.0f);
  x2h[(size_t)b * 64 + k] = (_Float16)fmaxf(e2, 0.0f);
  hfh[(size_t)b * 64 + k] = (_Float16)hidden[(size_t)b * 64 + k];
}

// =====================================================================
// K5: GRU elementwise combine.
// =====================================================================
__global__ void k5_gru(const float* __restrict__ gi, const float* __restrict__ gh,
                       const float* __restrict__ h, float* __restrict__ out,
                       _Float16* __restrict__ outh, int n) {
  int i = blockIdx.x * blockDim.x + threadIdx.x;
  if (i >= n) return;
  int b = i >> 6, k = i & 63;
  float ir = gi[(size_t)b * 192 + k], iz = gi[(size_t)b * 192 + 64 + k], in = gi[(size_t)b * 192 + 128 + k];
  float hr = gh[(size_t)b * 192 + k], hz = gh[(size_t)b * 192 + 64 + k], hn = gh[(size_t)b * 192 + 128 + k];
  float r = 1.0f / (1.0f + __expf(-(ir + hr)));
  float z = 1.0f / (1.0f + __expf(-(iz + hz)));
  float nn = tanhf(in + r * hn);
  float res = (1.0f - z) * nn + z * h[i];
  out[i] = res;
  if (outh) outh[i] = (_Float16)res;
}

// =====================================================================
// K6u: u[b,head,j] = sum_h hh[b,h]*heW2[j,1024+2h+head] + heW2[j,1152+head];
//      s[b,head]   = sum_h hh[b,h]*heb2[1024+2h+head]   + heb2[1152+head]
// =====================================================================
__global__ void k6_u(const float* __restrict__ hh, const float* __restrict__ heW2,
                     const float* __restrict__ heb2, float* __restrict__ u,
                     float* __restrict__ s) {
  int t = threadIdx.x;
  int b = blockIdx.x * 2 + (t >> 7);
  int idx = t & 127, head = idx >> 6, j = idx & 63;
  const float* hrow = hh + (size_t)b * 64;
  float acc = heW2[(size_t)j * HEOUT + 1152 + head];
  for (int h = 0; h < 64; ++h) acc += hrow[h] * heW2[(size_t)j * HEOUT + 1024 + 2 * h + head];
  u[((size_t)b * 2 + head) * 64 + j] = acc;
  if (j == 0) {
    float sa = heb2[1152 + head];
    for (int h = 0; h < 64; ++h) sa += hrow[h] * heb2[1024 + 2 * h + head];
    s[(size_t)b * 2 + head] = sa;
  }
}

// =====================================================================
// K6qa: q_attack per enemy row via hid_e . u + merger_out softmax.
// =====================================================================
__global__ void k6_qa(const float* __restrict__ hid_e, const float* __restrict__ u,
                      const float* __restrict__ s, const float* __restrict__ mow,
                      float* __restrict__ qatt, int n) {
  int er = blockIdx.x * blockDim.x + threadIdx.x;
  if (er >= n) return;
  int b = er >> 3;
  const float* hr = hid_e + (size_t)er * 64;
  const float* u0 = u + (size_t)b * 128;
  const float* u1 = u0 + 64;
  float q0 = s[b * 2], q1 = s[b * 2 + 1];
  for (int j = 0; j < 64; ++j) { float hv = hr[j]; q0 += hv * u0[j]; q1 += hv * u1[j]; }
  float m0 = mow[0], m1 = mow[1];
  float mx = fmaxf(m0, m1);
  float w0 = __expf(m0 - mx), w1 = __expf(m1 - mx);
  qatt[er] = (w0 * q0 + w1 * q1) / (w0 + w1);
}

// =====================================================================
// K7: local_q from g[:8] and per-agent MLP.
// =====================================================================
__global__ void k7_local(const float* __restrict__ g, const float* __restrict__ mlpW,
                         const float* __restrict__ mlpb, float* __restrict__ lq) {
  int t = threadIdx.x;
  if (t >= NAG * NACT) return;
  int a = t / NACT, k = t % NACT;
  float s = mlpb[a * NACT + k];
  for (int h = 0; h < 64; ++h) s += g[a * 64 + h] * mlpW[a * 64 * NACT + h * NACT + k];
  lq[t] = s;
}

// =====================================================================
// K8: final assembly: q_normal / q_attack / q_f / local -> outputs.
// =====================================================================
__global__ void k8_final(const float* __restrict__ hh, const float* __restrict__ z,
                         const float* __restrict__ qatt, const float* __restrict__ lq,
                         const float* __restrict__ fc2nW, const float* __restrict__ fc2nb,
                         const float* __restrict__ fc3W, const float* __restrict__ fc3b,
                         float* __restrict__ out_q, float* __restrict__ out_lqe,
                         float* __restrict__ out_qf, int n) {
  int i = blockIdx.x * blockDim.x + threadIdx.x;
  if (i >= n) return;
  int b = i / NACT, k = i % NACT;
  const float* zr = z + (size_t)b * 64;
  float qf = fc3b[k];
  for (int h = 0; h < 64; ++h) qf += zr[h] * fc3W[h * NACT + k];
  float qp;
  if (k < OUTN) {
    const float* hr = hh + (size_t)b * 64;
    qp = fc2nb[k];
    for (int h = 0; h < 64; ++h) qp += hr[h] * fc2nW[h * OUTN + k];
  } else {
    qp = qatt[b * 8 + (k - OUTN)];
  }
  float l = lq[(b & 7) * NACT + k];
  out_q[i]   = qp + l + qf;
  out_lqe[i] = l;
  out_qf[i]  = qf;
}

// =====================================================================
extern "C" void kernel_launch(void* const* d_in, const int* in_sizes, int n_in,
                              void* d_out, int out_size, void* d_ws, size_t ws_size,
                              hipStream_t stream) {
  (void)in_sizes; (void)n_in; (void)out_size; (void)ws_size;
  const float* own      = (const float*)d_in[1];
  const float* ef       = (const float*)d_in[2];
  const float* af       = (const float*)d_in[3];
  const int*   aids     = (const int*)  d_in[4];
  const int*   lids     = (const int*)  d_in[5];
  const float* hidden   = (const float*)d_in[6];
  const float* fc1W     = (const float*)d_in[7];
  const float* fc1b     = (const float*)d_in[8];
  const float* aemb     = (const float*)d_in[9];
  const float* acemb    = (const float*)d_in[10];
  const float* heW1     = (const float*)d_in[11];
  const float* heb1     = (const float*)d_in[12];
  const float* heW2     = (const float*)d_in[13];
  const float* heb2     = (const float*)d_in[14];
  const float* haW1     = (const float*)d_in[15];
  const float* hab1     = (const float*)d_in[16];
  const float* haW2     = (const float*)d_in[17];
  const float* hab2     = (const float*)d_in[18];
  const float* mwin     = (const float*)d_in[19];
  const float* mow      = (const float*)d_in[20];
  const float* g1Wih    = (const float*)d_in[21];
  const float* g1Whh    = (const float*)d_in[22];
  const float* g1bih    = (const float*)d_in[23];
  const float* g1bhh    = (const float*)d_in[24];
  const float* fc2nW    = (const float*)d_in[25];
  const float* fc2nb    = (const float*)d_in[26];
  const float* fc1bW    = (const float*)d_in[27];
  const float* fc1bb    = (const float*)d_in[28];
  const float* g2Wih    = (const float*)d_in[29];
  const float* g2Whh    = (const float*)d_in[30];
  const float* g2bih    = (const float*)d_in[31];
  const float* g2bhh    = (const float*)d_in[32];
  const float* encW1    = (const float*)d_in[33];
  const float* encb1    = (const float*)d_in[34];
  const float* encW2    = (const float*)d_in[35];
  const float* encb2    = (const float*)d_in[36];
  const float* fc3W     = (const float*)d_in[37];
  const float* fc3b     = (const float*)d_in[38];
  const float* mlpW     = (const float*)d_in[39];
  const float* mlpb     = (const float*)d_in[40];

  char* ws = (char*)d_ws;
  float*    hid_e  = (float*)   (ws + OFF_HIDE);
  _Float16* Aws    = (_Float16*)(ws + OFF_AWS);
  _Float16* Wcat   = (_Float16*)(ws + OFF_WCAT);
  _Float16* Wg1i   = (_Float16*)(ws + OFF_WG1I);
  _Float16* Wg1h   = (_Float16*)(ws + OFF_WG1H);
  _Float16* Wg2i   = (_Float16*)(ws + OFF_WG2I);
  _Float16* Wg2h   = (_Float16*)(ws + OFF_WG2H);
  _Float16* We1    = (_Float16*)(ws + OFF_WE1);
  _Float16* We2    = (_Float16*)(ws + OFF_WE2);
  float*    embsum = (float*)   (ws + OFF_EMB);
  _Float16* xh     = (_Float16*)(ws + OFF_XH);
  _Float16* x2h    = (_Float16*)(ws + OFF_X2H);
  _Float16* hfh    = (_Float16*)(ws + OFF_HFH);
  float*    gi1    = (float*)   (ws + OFF_GI1);
  float*    gh1    = (float*)   (ws + OFF_GH1);
  float*    gi2    = (float*)   (ws + OFF_GI2);
  float*    gh2    = (float*)   (ws + OFF_GH2);
  float*    gbuf   = (float*)   (ws + OFF_G);
  _Float16* gh16   = (_Float16*)(ws + OFF_GH16);
  _Float16* z1h    = (_Float16*)(ws + OFF_Z1H);
  float*    zbuf   = (float*)   (ws + OFF_Z);
  float*    ubuf   = (float*)   (ws + OFF_U);
  float*    sbuf   = (float*)   (ws + OFF_S);
  float*    qatt   = (float*)   (ws + OFF_QATT);
  float*    lq     = (float*)   (ws + OFF_LQ);

  float* out     = (float*)d_out;
  float* out_q   = out;                                  // (bs,8,14)
  float* out_hh  = out + (size_t)Btot * NACT;            // (bs,8,64)
  float* out_lqe = out_hh + (size_t)Btot * H;            // (bs,8,14)
  float* out_qf  = out_lqe + (size_t)Btot * NACT;        // (bs,8,14)

  // 0) weight prep
  k0_prep<<<256, 256, 0, stream>>>(heW2, heb2, haW2, hab2, g1Wih, g1Whh, g2Wih, g2Whh,
                                   encW1, encW2, Wcat, Wg1i, Wg1h, Wg2i, Wg2h, We1, We2);
  // 1) hypernet layer-1 + G/S
  k1_enemy<<<Btot / 16, 256, 0, stream>>>(ef, heW1, heb1, hid_e, Aws);
  k1_ally <<<Btot / 16, 256, 0, stream>>>(af, haW1, hab1, Aws);
  // 2) embsum = [Ge|Ga|Se|Sa] @ Wcat^T  -> (B,128)
  gemm_wmma_f16<<<(Btot / 16) * (128 / 16) / 8, 256, 0, stream>>>(
      Aws, Wcat, nullptr, embsum, nullptr, Btot, 128, KCAT, 0);
  // 3) agent fuse -> x, x2, h (f16)
  k3_agent<<<Btot / 4, 256, 0, stream>>>(own, aids, lids, hidden, fc1W, fc1b, aemb, acemb,
                                         mwin, embsum, fc1bW, fc1bb, xh, x2h, hfh);
  // 4) GRU gate GEMMs (N=192, K=64)
  int gruBlocks = (Btot / 16) * (192 / 16) / 8;
  gemm_wmma_f16<<<gruBlocks, 256, 0, stream>>>(xh,  Wg1i, g1bih, gi1, nullptr, Btot, 192, 64, 0);
  gemm_wmma_f16<<<gruBlocks, 256, 0, stream>>>(hfh, Wg1h, g1bhh, gh1, nullptr, Btot, 192, 64, 0);
  gemm_wmma_f16<<<gruBlocks, 256, 0, stream>>>(x2h, Wg2i, g2bih, gi2, nullptr, Btot, 192, 64, 0);
  gemm_wmma_f16<<<gruBlocks, 256, 0, stream>>>(hfh, Wg2h, g2bhh, gh2, nullptr, Btot, 192, 64, 0);
  // 5) GRU combines
  k5_gru<<<Btot * H / 256, 256, 0, stream>>>(gi1, gh1, hidden, out_hh, nullptr, Btot * H);
  k5_gru<<<Btot * H / 256, 256, 0, stream>>>(gi2, gh2, hidden, gbuf, gh16, Btot * H);
  // 6) encoder GEMMs
  gemm_wmma_f16<<<(Btot / 16) * (128 / 16) / 8, 256, 0, stream>>>(
      gh16, We1, encb1, nullptr, z1h, Btot, 128, 64, 1);
  gemm_wmma_f16<<<(Btot / 16) * (64 / 16) / 8, 256, 0, stream>>>(
      z1h, We2, encb2, zbuf, nullptr, Btot, 64, 128, 0);
  // 7) attack-head u/s, qa
  k6_u<<<Btot / 2, 256, 0, stream>>>(out_hh, heW2, heb2, ubuf, sbuf);
  k6_qa<<<RE / 256, 256, 0, stream>>>(hid_e, ubuf, sbuf, mow, qatt, RE);
  // 8) local_q and final assembly
  k7_local<<<1, 128, 0, stream>>>(gbuf, mlpW, mlpb, lq);
  int nfin = Btot * NACT;
  k8_final<<<(nfin + 255) / 256, 256, 0, stream>>>(out_hh, zbuf, qatt, lq, fc2nW, fc2nb,
                                                   fc3W, fc3b, out_q, out_lqe, out_qf, nfin);
}
